// DeepLabHeadV3Plus_19000935318116
// MI455X (gfx1250) — matrix operand order repaired and checked by reading, not compiled, not run on hardware
//
#include <hip/hip_runtime.h>

// ---------------------------------------------------------------------------
// DeepLabV3+ head, MI455X (gfx1250, wave32).
// Matrix math: v_wmma_f32_16x16x32_bf16, 2x2 register-blocked (32x32 C tile
// per wave: 4 WMMAs per 8 b128 loads), branch-free fragment loads.
// Layouts:  weights  [Mp][Kp] bf16 row-major  (Mp % 32, Kp % 32, zero padded)
//           activations NHWC bf16 [B][Np][Cp] (Np % 32 or exact, Cp % 32)
// ---------------------------------------------------------------------------

typedef __attribute__((ext_vector_type(16))) __bf16 v16bf;
typedef __attribute__((ext_vector_type(8)))  __bf16 v8bf;
typedef __attribute__((ext_vector_type(8)))  float  v8f;

#define BB   4      // batch
#define SPK  196    // superpixels per image
#define SPHW 224    // superpixel map side

__device__ __forceinline__ __bf16 to_bf(float f) { return (__bf16)f; }

// A fragment (16x32 bf16): 16 consecutive K values
__device__ __forceinline__ v16bf ld_frag_a(const __bf16* __restrict__ p) {
  v8bf x = *(const v8bf*)p;
  v8bf y = *(const v8bf*)(p + 8);
  return __builtin_shufflevector(x, y, 0,1,2,3,4,5,6,7,8,9,10,11,12,13,14,15);
}
// B fragment (32x16 bf16, B stored transposed): 16 consecutive K values
__device__ __forceinline__ v16bf ld_frag_b(const __bf16* __restrict__ p) {
  v8bf x = *(const v8bf*)p;
  v8bf y = *(const v8bf*)(p + 8);
  return __builtin_shufflevector(x, y, 0,1,2,3,4,5,6,7,8,9,10,11,12,13,14,15);
}

// ------------------------------ packing ------------------------------------
__global__ void zero_k(float* __restrict__ p, int n) {
  int i = blockIdx.x * blockDim.x + threadIdx.x;
  if (i < n) p[i] = 0.0f;
}

// NCHW f32 -> NHWC bf16, pixel rows padded to Np (pad rows zeroed)
__global__ void pack_nhwc_k(const float* __restrict__ src, __bf16* __restrict__ dst,
                            int C, int HW, int Np, int Bn) {
  int idx = blockIdx.x * blockDim.x + threadIdx.x;
  int total = Bn * Np * C;
  if (idx >= total) return;
  int c = idx % C;
  int p = (idx / C) % Np;
  int b = idx / (C * Np);
  dst[idx] = (p < HW) ? to_bf(src[((size_t)b * C + c) * HW + p]) : to_bf(0.0f);
}

// [Cout][Cin][3][3] f32 -> tap-major [Coutp][9][Cinp] bf16 (zero padded)
__global__ void pack_w3x3_k(const float* __restrict__ src, __bf16* __restrict__ dst,
                            int Cout, int Cin, int Cinp, int Coutp) {
  int idx = blockIdx.x * blockDim.x + threadIdx.x;
  int total = Coutp * 9 * Cinp;
  if (idx >= total) return;
  int ci = idx % Cinp;
  int r  = (idx / Cinp) % 9;
  int co = idx / (Cinp * 9);
  __bf16 v = to_bf(0.0f);
  if (co < Cout && ci < Cin) v = to_bf(src[((size_t)co * Cin + ci) * 9 + r]);
  dst[idx] = v;
}

// [Cout][K] f32 -> [Coutp][K] bf16 (rows >= Cout zeroed); K % 32 == 0
__global__ void pack_w1x1_k(const float* __restrict__ src, __bf16* __restrict__ dst,
                            int Cout, int K, int Coutp) {
  int idx = blockIdx.x * blockDim.x + threadIdx.x;
  if (idx >= Coutp * K) return;
  int co = idx / K;
  dst[idx] = (co < Cout) ? to_bf(src[idx]) : to_bf(0.0f);
}

// center tap of [Cout][Cin][3][3] -> [Cout][Cin] bf16 (dilated-conv degeneracy)
__global__ void pack_wcenter_k(const float* __restrict__ src, __bf16* __restrict__ dst, int n) {
  int i = blockIdx.x * blockDim.x + threadIdx.x;
  if (i < n) dst[i] = to_bf(src[(size_t)i * 9 + 4]);
}

// global average pool of feat_out -> poolT [32][2048] bf16 (rows 4..31 zero)
__global__ void gap_pool_k(const float* __restrict__ x, __bf16* __restrict__ poolT) {
  int idx = blockIdx.x * blockDim.x + threadIdx.x;
  if (idx >= 32 * 2048) return;
  int n = idx / 2048, k = idx % 2048;
  float s = 0.0f;
  if (n < BB) {
    const float* p = x + ((size_t)n * 2048 + k) * 196;
    for (int j = 0; j < 196; ++j) s += p[j];
    s *= (1.0f / 196.0f);
  }
  poolT[idx] = to_bf(s);
}

// ------------------------------ WMMA GEMM ----------------------------------
// C[b](MxN,f32) = A(Mp x K) * BT[b](Np x K)^T.  32x32 macro-tile per wave:
// 2 A frags + 2 B frags -> 4 WMMAs per 32-K step (8 MAC/byte of operands).
__global__ void gemm_bf16_wmma(const __bf16* __restrict__ A, const __bf16* __restrict__ BT,
                               float* __restrict__ C, int M, int N, int K,
                               long long strideBT, long long strideC) {
  const __bf16* Bp = BT + (size_t)blockIdx.z * strideBT;
  float* Cp = C + (size_t)blockIdx.z * strideC;
  int lane = threadIdx.x & 31;
  int wave = threadIdx.x >> 5;
  int Mt = (M + 31) >> 5, Nt = (N + 31) >> 5;
  int tile = blockIdx.x * (blockDim.x >> 5) + wave;
  if (tile >= Mt * Nt) return;                 // wave-uniform: EXEC all-1s at WMMA
  int m0 = (tile % Mt) << 5;
  int n0 = (tile / Mt) << 5;
  int hi = lane >> 4;
  int l15 = lane & 15;
  const __bf16* ar0 = A  + (size_t)(m0 + l15) * K + hi * 16;
  const __bf16* ar1 = ar0 + (size_t)16 * K;
  const __bf16* br0 = Bp + (size_t)(n0 + l15) * K + hi * 16;
  const __bf16* br1 = br0 + (size_t)16 * K;
  v8f acc00 = {}, acc01 = {}, acc10 = {}, acc11 = {};
  for (int k0 = 0; k0 < K; k0 += 32) {
    v16bf a0 = ld_frag_a(ar0 + k0);
    v16bf a1 = ld_frag_a(ar1 + k0);
    v16bf b0 = ld_frag_b(br0 + k0);
    v16bf b1 = ld_frag_b(br1 + k0);
    if (k0 + 32 < K) {
      __builtin_prefetch(ar0 + k0 + 32, 0, 1);
      __builtin_prefetch(br0 + k0 + 32, 0, 1);
    }
    acc00 = __builtin_amdgcn_wmma_f32_16x16x32_bf16(false, a0, false, b0, (short)0, acc00, false, false);
    acc01 = __builtin_amdgcn_wmma_f32_16x16x32_bf16(false, a0, false, b1, (short)0, acc01, false, false);
    acc10 = __builtin_amdgcn_wmma_f32_16x16x32_bf16(false, a1, false, b0, (short)0, acc10, false, false);
    acc11 = __builtin_amdgcn_wmma_f32_16x16x32_bf16(false, a1, false, b1, (short)0, acc11, false, false);
  }
  int nc0 = n0 + l15, nc1 = n0 + 16 + l15;
#pragma unroll
  for (int v = 0; v < 8; ++v) {
    int ma = m0 + v + (hi << 3);
    int mb = ma + 16;
    if (ma < M && nc0 < N) Cp[(size_t)ma * N + nc0] = acc00[v];
    if (ma < M && nc1 < N) Cp[(size_t)ma * N + nc1] = acc01[v];
    if (mb < M && nc0 < N) Cp[(size_t)mb * N + nc0] = acc10[v];
    if (mb < M && nc1 < N) Cp[(size_t)mb * N + nc1] = acc11[v];
  }
}

// --------------------- WMMA implicit-GEMM 3x3 conv -------------------------
// pad == dil.  W tap-major [Coutp][9*Cinp]; X NHWC [B][HW][Cinp].
// 32x32 macro-tile per wave; each 16-K chunk sits inside one tap ->
// contiguous NHWC load; halo = per-lane pointer select onto a zero page.
__global__ void conv3x3_bf16_wmma(const __bf16* __restrict__ W, const __bf16* __restrict__ X,
                                  const __bf16* __restrict__ zpage,
                                  float* __restrict__ C, int Cout, int Cinp, int H, int dil,
                                  long long strideX) {
  int HW = H * H;
  int K = 9 * Cinp;
  const __bf16* Xp = X + (size_t)blockIdx.z * strideX;
  float* Cp = C + (size_t)blockIdx.z * Cout * HW;
  int lane = threadIdx.x & 31;
  int wave = threadIdx.x >> 5;
  int Mt = (Cout + 31) >> 5, Nt = (HW + 31) >> 5;
  int tile = blockIdx.x * (blockDim.x >> 5) + wave;
  if (tile >= Mt * Nt) return;
  int m0 = (tile % Mt) << 5;
  int n0 = (tile / Mt) << 5;
  int hi = lane >> 4;
  int l15 = lane & 15;
  int nc0 = n0 + l15, nc1 = n0 + 16 + l15;
  bool pv0 = nc0 < HW, pv1 = nc1 < HW;
  int h0 = pv0 ? (nc0 / H) : 0, w0 = pv0 ? (nc0 - h0 * H) : 0;
  int h1 = pv1 ? (nc1 / H) : 0, w1 = pv1 ? (nc1 - h1 * H) : 0;
  const __bf16* ar0 = W + (size_t)(m0 + l15) * K + hi * 16;
  const __bf16* ar1 = ar0 + (size_t)16 * K;
  v8f acc00 = {}, acc01 = {}, acc10 = {}, acc11 = {};
  int tap = 0;           // chunk start kc = tap*Cinp + ci0  (== k0 + hi*16)
  int ci0 = hi * 16;
  for (int k0 = 0; k0 < K; k0 += 32) {
    int ky = tap / 3, kx = tap - ky * 3;
    int dy = (ky - 1) * dil, dx = (kx - 1) * dil;
    int ih0 = h0 + dy, iw0 = w0 + dx;
    int ih1 = h1 + dy, iw1 = w1 + dx;
    bool ok0 = pv0 && (ih0 >= 0) && (ih0 < H) && (iw0 >= 0) && (iw0 < H);
    bool ok1 = pv1 && (ih1 >= 0) && (ih1 < H) && (iw1 >= 0) && (iw1 < H);
    const __bf16* bp0 = ok0 ? (Xp + (size_t)(ih0 * H + iw0) * Cinp + ci0) : zpage;
    const __bf16* bp1 = ok1 ? (Xp + (size_t)(ih1 * H + iw1) * Cinp + ci0) : zpage;
    v16bf a0 = ld_frag_a(ar0 + k0);
    v16bf a1 = ld_frag_a(ar1 + k0);
    v16bf b0 = ld_frag_b(bp0);
    v16bf b1 = ld_frag_b(bp1);
    acc00 = __builtin_amdgcn_wmma_f32_16x16x32_bf16(false, a0, false, b0, (short)0, acc00, false, false);
    acc01 = __builtin_amdgcn_wmma_f32_16x16x32_bf16(false, a0, false, b1, (short)0, acc01, false, false);
    acc10 = __builtin_amdgcn_wmma_f32_16x16x32_bf16(false, a1, false, b0, (short)0, acc10, false, false);
    acc11 = __builtin_amdgcn_wmma_f32_16x16x32_bf16(false, a1, false, b1, (short)0, acc11, false, false);
    ci0 += 32;
    if (ci0 >= Cinp) { ci0 -= Cinp; ++tap; }
  }
#pragma unroll
  for (int v = 0; v < 8; ++v) {
    int ma = m0 + v + (hi << 3);
    int mb = ma + 16;
    if (ma < Cout && pv0) Cp[(size_t)ma * HW + nc0] = acc00[v];
    if (ma < Cout && pv1) Cp[(size_t)ma * HW + nc1] = acc01[v];
    if (mb < Cout && pv0) Cp[(size_t)mb * HW + nc0] = acc10[v];
    if (mb < Cout && pv1) Cp[(size_t)mb * HW + nc1] = acc11[v];
  }
}

// ------------------------- BatchNorm (training) ----------------------------
__global__ void bn_stats_k(const float* __restrict__ x, float* __restrict__ mean,
                           float* __restrict__ rvar, int Cc, int HW, int Bn) {
  int c = blockIdx.x;
  __shared__ float ssum[256], ssq[256];
  float s = 0.0f, s2 = 0.0f;
  for (int b = 0; b < Bn; ++b) {
    const float* p = x + ((size_t)b * Cc + c) * HW;
    for (int i = threadIdx.x; i < HW; i += blockDim.x) {
      float v = p[i]; s += v; s2 += v * v;
    }
  }
  ssum[threadIdx.x] = s; ssq[threadIdx.x] = s2;
  __syncthreads();
  for (int st = blockDim.x >> 1; st > 0; st >>= 1) {
    if ((int)threadIdx.x < st) {
      ssum[threadIdx.x] += ssum[threadIdx.x + st];
      ssq[threadIdx.x]  += ssq[threadIdx.x + st];
    }
    __syncthreads();
  }
  if (threadIdx.x == 0) {
    float inv = 1.0f / (float)(Bn * HW);
    float m = ssum[0] * inv;
    float var = ssq[0] * inv - m * m;
    mean[c] = m;
    rvar[c] = rsqrtf(var + 1e-5f);
  }
}

// normalize + ReLU.  outf: NCHW f32 (channel-concat via CfTot/cOffF).
// outb: NHWC bf16 (row length Kb, rows NpB, channel offset cOffB).
__global__ void bn_relu_k(const float* __restrict__ x, const float* __restrict__ mean,
                          const float* __restrict__ rvar, const float* __restrict__ g,
                          const float* __restrict__ bt, int Cc, int HW, int Bn,
                          float* __restrict__ outf, int CfTot, int cOffF,
                          __bf16* __restrict__ outb, int Kb, int NpB, int cOffB) {
  int idx = blockIdx.x * blockDim.x + threadIdx.x;
  int total = Bn * Cc * HW;
  if (idx >= total) return;
  int i = idx % HW;
  int c = (idx / HW) % Cc;
  int b = idx / (HW * Cc);
  float y = (x[idx] - mean[c]) * rvar[c] * g[c] + bt[c];
  y = fmaxf(y, 0.0f);
  if (outf) outf[((size_t)b * CfTot + cOffF + c) * HW + i] = y;
  if (outb) outb[((size_t)b * NpB + i) * Kb + cOffB + c] = to_bf(y);
}

// pool branch: BN over the 4 batch samples, ReLU, broadcast into NHWC concat
__global__ void gp_bn_broadcast_k(const float* __restrict__ gp, const float* __restrict__ g,
                                  const float* __restrict__ bt, __bf16* __restrict__ concat) {
  int c = threadIdx.x;  // 256 threads; gp is [256][4]
  float v[BB];
  float m = 0.0f;
  for (int b = 0; b < BB; ++b) { v[b] = gp[(size_t)c * BB + b]; m += v[b]; }
  m *= (1.0f / BB);
  float var = 0.0f;
  for (int b = 0; b < BB; ++b) { float d = v[b] - m; var += d * d; }
  var *= (1.0f / BB);
  float rv = rsqrtf(var + 1e-5f);
  for (int b = 0; b < BB; ++b) {
    float y = fmaxf((v[b] - m) * rv * g[c] + bt[c], 0.0f);
    __bf16 yb = to_bf(y);
    for (int p = 0; p < 196; ++p)
      concat[((size_t)b * 224 + p) * 1280 + 1024 + c] = yb;
  }
}

// ------------------------------ bilinear -----------------------------------
__device__ __forceinline__ float bilin(const float* __restrict__ img, int Hin,
                                       float fy, float fx) {
  fy = fminf(fmaxf(fy, 0.0f), (float)(Hin - 1));
  fx = fminf(fmaxf(fx, 0.0f), (float)(Hin - 1));
  int y0 = (int)fy, x0 = (int)fx;
  int y1 = min(y0 + 1, Hin - 1), x1 = min(x0 + 1, Hin - 1);
  float wy = fy - (float)y0, wx = fx - (float)x0;
  float a = img[y0 * Hin + x0], b = img[y0 * Hin + x1];
  float c = img[y1 * Hin + x0], d = img[y1 * Hin + x1];
  float top = a + (b - a) * wx;
  float bot = c + (d - c) * wx;
  return top + (bot - top) * wy;
}

// ASPP 14x14 -> 56x56 into ffea channels 48..303 (f32 NCHW + bf16 NHWC)
__global__ void resize14to56_k(const float* __restrict__ src, float* __restrict__ ffea_f,
                               __bf16* __restrict__ ffea_b) {
  int idx = blockIdx.x * blockDim.x + threadIdx.x;
  const int total = BB * 256 * 3136;
  if (idx >= total) return;
  int i = idx % 3136;
  int c = (idx / 3136) % 256;
  int b = idx / (3136 * 256);
  int h = i / 56, w = i % 56;
  const float* img = src + ((size_t)b * 256 + c) * 196;
  float y = bilin(img, 14, (h + 0.5f) * 0.25f - 0.5f, (w + 0.5f) * 0.25f - 0.5f);
  ffea_f[((size_t)b * 304 + 48 + c) * 3136 + i] = y;
  ffea_b[((size_t)b * 3136 + i) * 320 + 48 + c] = to_bf(y);
}

__global__ void bias_add_k(const float* __restrict__ src, const float* __restrict__ bias,
                           int Cc, int HW, int total, float* __restrict__ dst) {
  int idx = blockIdx.x * blockDim.x + threadIdx.x;
  if (idx >= total) return;
  dst[idx] = src[idx] + bias[(idx / HW) % Cc];
}

// -------------------- fused resize(56->224) + segment-sum ------------------
// Reference does raw CHW->HWC reshape of ffea224 then segment-means.
// Invert the flat index -> (c,h,w) and bilinearly sample ffea on the fly.
__global__ void sp_accum_k(const float* __restrict__ ffea_f, const int* __restrict__ sp,
                           float* __restrict__ sums, float* __restrict__ counts) {
  int idx = blockIdx.x * blockDim.x + threadIdx.x;
  const int perB = SPHW * SPHW;  // 50176
  if (idx >= BB * perB) return;
  int b = idx / perB;
  int r = idx - b * perB;
  int lab = sp[idx] + b * SPK;
  atomicAdd(&counts[lab], 1.0f);
  const float* fb = ffea_f + (size_t)b * 304 * 3136;
  float* srow = sums + (size_t)lab * 304;
  long long f = (long long)r * 304;
  for (int cp = 0; cp < 304; ++cp, ++f) {
    int c = (int)(f / perB);
    int rem = (int)(f - (long long)c * perB);
    int h = rem / SPHW;
    int w = rem - h * SPHW;
    float v = bilin(fb + (size_t)c * 3136, 56,
                    (h + 0.5f) * 0.25f - 0.5f, (w + 0.5f) * 0.25f - 0.5f);
    atomicAdd(&srow[cp], v);
  }
}

__global__ void local_cls_k(const float* __restrict__ sums, const float* __restrict__ counts,
                            const float* __restrict__ lw, const float* __restrict__ lb,
                            float* __restrict__ out) {
  int n = blockIdx.x;    // 784
  int j = threadIdx.x;   // 64
  float inv = 1.0f / fmaxf(counts[n], 1.0f);
  const float* srow = sums + (size_t)n * 304;
  const float* wrow = lw + (size_t)j * 304;
  float s = lb[j];
  for (int c = 0; c < 304; ++c) s += srow[c] * inv * wrow[c];
  out[n * 64 + j] = s;
}

// ---------------------------------------------------------------------------
extern "C" void kernel_launch(void* const* d_in, const int* in_sizes, int n_in,
                              void* d_out, int out_size, void* d_ws, size_t ws_size,
                              hipStream_t stream) {
  (void)in_sizes; (void)n_in; (void)out_size; (void)ws_size;

  const float* feat_out = (const float*)d_in[0];   // [4,2048,14,14]
  const float* feat_low = (const float*)d_in[1];   // [4,256,56,56]
  const int*   spmap    = (const int*)  d_in[2];   // [4,224,224]
  const float* w_low  = (const float*)d_in[3];
  const float* g_low  = (const float*)d_in[4];
  const float* b_low  = (const float*)d_in[5];
  const float* w_a0   = (const float*)d_in[6];
  const float* g_a0   = (const float*)d_in[7];
  const float* b_a0   = (const float*)d_in[8];
  const float* w_a1   = (const float*)d_in[9];
  const float* g_a1   = (const float*)d_in[10];
  const float* b_a1   = (const float*)d_in[11];
  const float* w_a2   = (const float*)d_in[12];
  const float* g_a2   = (const float*)d_in[13];
  const float* b_a2   = (const float*)d_in[14];
  const float* w_a3   = (const float*)d_in[15];
  const float* g_a3   = (const float*)d_in[16];
  const float* b_a3   = (const float*)d_in[17];
  const float* w_pool = (const float*)d_in[18];
  const float* g_pool = (const float*)d_in[19];
  const float* b_pool = (const float*)d_in[20];
  const float* w_proj = (const float*)d_in[21];
  const float* g_proj = (const float*)d_in[22];
  const float* b_proj = (const float*)d_in[23];
  const float* w_cls1 = (const float*)d_in[24];
  const float* g_cls1 = (const float*)d_in[25];
  const float* b_cls1 = (const float*)d_in[26];
  const float* w_cls2 = (const float*)d_in[27];
  const float* cls2_bias = (const float*)d_in[28];
  const float* w_rec1 = (const float*)d_in[29];
  const float* g_rec1 = (const float*)d_in[30];
  const float* b_rec1 = (const float*)d_in[31];
  const float* w_rec2 = (const float*)d_in[32];
  const float* rec2_bias = (const float*)d_in[33];
  const float* lin_w  = (const float*)d_in[34];
  const float* lin_b  = (const float*)d_in[35];

  // ---- workspace carve-up ----
  char* base = (char*)d_ws;
  size_t off = 0;
  auto alloc = [&](size_t bytes) -> void* {
    void* p = base + off;
    off = (off + bytes + 255) & ~(size_t)255;
    return p;
  };
  // bf16 (padded) operand buffers
  __bf16* wb_low  = (__bf16*)alloc((size_t)64 * 256 * 2);            // M 48->64
  __bf16* wb_a0   = (__bf16*)alloc((size_t)256 * 2048 * 2);
  __bf16* wb_a1   = (__bf16*)alloc((size_t)256 * 9 * 2048 * 2);      // tap-major
  __bf16* wb_a2c  = (__bf16*)alloc((size_t)256 * 2048 * 2);
  __bf16* wb_a3c  = (__bf16*)alloc((size_t)256 * 2048 * 2);
  __bf16* wb_pool = (__bf16*)alloc((size_t)256 * 2048 * 2);
  __bf16* wb_proj = (__bf16*)alloc((size_t)256 * 1280 * 2);
  __bf16* wb_cls1 = (__bf16*)alloc((size_t)256 * 9 * 320 * 2);       // Cin 304->320
  __bf16* wb_cls2 = (__bf16*)alloc((size_t)32 * 256 * 2);            // M 1->32
  __bf16* wb_rec1 = (__bf16*)alloc((size_t)256 * 9 * 320 * 2);
  __bf16* wb_rec2 = (__bf16*)alloc((size_t)32 * 256 * 2);            // M 3->32
  __bf16* xb_out  = (__bf16*)alloc((size_t)BB * 224 * 2048 * 2);     // NHWC, 196->224
  __bf16* xb_low  = (__bf16*)alloc((size_t)BB * 3136 * 256 * 2);     // NHWC
  __bf16* poolT   = (__bf16*)alloc((size_t)32 * 2048 * 2);           // N 4->32
  __bf16* concat_b= (__bf16*)alloc((size_t)BB * 224 * 1280 * 2);     // NHWC
  __bf16* ffea_b  = (__bf16*)alloc((size_t)BB * 3136 * 320 * 2);     // NHWC, C 304->320
  __bf16* segact_b= (__bf16*)alloc((size_t)BB * 3136 * 256 * 2);     // NHWC
  __bf16* recact_b= (__bf16*)alloc((size_t)BB * 3136 * 256 * 2);
  __bf16* zpage   = (__bf16*)alloc(256);                             // 128 zeros
  // f32 stage buffers (NCHW)
  float* low_raw  = (float*)alloc((size_t)BB * 48 * 3136 * 4);
  float* b0_raw   = (float*)alloc((size_t)BB * 256 * 196 * 4);
  float* b1_raw   = (float*)alloc((size_t)BB * 256 * 196 * 4);
  float* b2_raw   = (float*)alloc((size_t)BB * 256 * 196 * 4);
  float* b3_raw   = (float*)alloc((size_t)BB * 256 * 196 * 4);
  float* gp_raw   = (float*)alloc((size_t)256 * BB * 4);             // [256][4]
  float* proj_raw = (float*)alloc((size_t)BB * 256 * 196 * 4);
  float* aspp_act = (float*)alloc((size_t)BB * 256 * 196 * 4);
  float* ffea_f   = (float*)alloc((size_t)BB * 304 * 3136 * 4);
  float* seg_raw  = (float*)alloc((size_t)BB * 256 * 3136 * 4);
  float* rec_raw  = (float*)alloc((size_t)BB * 256 * 3136 * 4);
  float* cls2_raw = (float*)alloc((size_t)BB * 1 * 3136 * 4);
  float* rec2_raw = (float*)alloc((size_t)BB * 3 * 3136 * 4);
  float* mean_buf = (float*)alloc((size_t)256 * 4);
  float* rvar_buf = (float*)alloc((size_t)256 * 4);
  float* sums     = (float*)alloc((size_t)BB * SPK * 304 * 4);
  float* counts   = (float*)alloc((size_t)BB * SPK * 4);

  auto g1 = [](int n) { return dim3((unsigned)((n + 255) / 256)); };
  auto zeroBF = [&](__bf16* p, size_t elems) {   // elems is even
    zero_k<<<g1((int)(elems / 2)), 256, 0, stream>>>((float*)p, (int)(elems / 2));
  };
  auto gemm = [&](const __bf16* A, const __bf16* BT, float* C,
                  int M, int N, int K, int Np, int batch) {
    int Mt = (M + 31) / 32, Nt = (N + 31) / 32;
    int blocks = (Mt * Nt + 3) / 4;
    gemm_bf16_wmma<<<dim3(blocks, 1, batch), 128, 0, stream>>>(
        A, BT, C, M, N, K, (long long)Np * K, (long long)M * N);
  };
  auto conv3 = [&](const __bf16* W, const __bf16* X, float* C,
                   int Cout, int Cinp, int H, int dil, long long strideX) {
    int HW = H * H;
    int Mt = (Cout + 31) / 32, Nt = (HW + 31) / 32;
    int blocks = (Mt * Nt + 3) / 4;
    conv3x3_bf16_wmma<<<dim3(blocks, 1, BB), 128, 0, stream>>>(
        W, X, zpage, C, Cout, Cinp, H, dil, strideX);
  };
  auto bn = [&](const float* raw, const float* g, const float* bt, int Cc, int HW,
                float* outf, int CfTot, int cOffF,
                __bf16* outb, int Kb, int NpB, int cOffB) {
    bn_stats_k<<<dim3(Cc), 256, 0, stream>>>(raw, mean_buf, rvar_buf, Cc, HW, BB);
    int total = BB * Cc * HW;
    bn_relu_k<<<g1(total), 256, 0, stream>>>(raw, mean_buf, rvar_buf, g, bt, Cc, HW, BB,
                                             outf, CfTot, cOffF, outb, Kb, NpB, cOffB);
  };

  // ---- 1) zero pads + pack operands to bf16 ----
  zeroBF(zpage, 128);
  zeroBF(concat_b, (size_t)BB * 224 * 1280);   // pad pixel rows 196..223
  zeroBF(ffea_b,   (size_t)BB * 3136 * 320);   // pad channels 304..319
  zero_k<<<g1(BB * SPK * 304), 256, 0, stream>>>(sums, BB * SPK * 304);
  zero_k<<<g1(BB * SPK), 256, 0, stream>>>(counts, BB * SPK);

  pack_w1x1_k<<<g1(64 * 256), 256, 0, stream>>>(w_low, wb_low, 48, 256, 64);
  pack_w1x1_k<<<g1(256 * 2048), 256, 0, stream>>>(w_a0, wb_a0, 256, 2048, 256);
  pack_w3x3_k<<<g1(256 * 9 * 2048), 256, 0, stream>>>(w_a1, wb_a1, 256, 2048, 2048, 256);
  pack_wcenter_k<<<g1(256 * 2048), 256, 0, stream>>>(w_a2, wb_a2c, 256 * 2048); // dil=24: center only
  pack_wcenter_k<<<g1(256 * 2048), 256, 0, stream>>>(w_a3, wb_a3c, 256 * 2048); // dil=36: center only
  pack_w1x1_k<<<g1(256 * 2048), 256, 0, stream>>>(w_pool, wb_pool, 256, 2048, 256);
  pack_w1x1_k<<<g1(256 * 1280), 256, 0, stream>>>(w_proj, wb_proj, 256, 1280, 256);
  pack_w3x3_k<<<g1(256 * 9 * 320), 256, 0, stream>>>(w_cls1, wb_cls1, 256, 304, 320, 256);
  pack_w1x1_k<<<g1(32 * 256), 256, 0, stream>>>(w_cls2, wb_cls2, 1, 256, 32);
  pack_w3x3_k<<<g1(256 * 9 * 320), 256, 0, stream>>>(w_rec1, wb_rec1, 256, 304, 320, 256);
  pack_w1x1_k<<<g1(32 * 256), 256, 0, stream>>>(w_rec2, wb_rec2, 3, 256, 32);
  pack_nhwc_k<<<g1(BB * 224 * 2048), 256, 0, stream>>>(feat_out, xb_out, 2048, 196, 224, BB);
  pack_nhwc_k<<<g1(BB * 3136 * 256), 256, 0, stream>>>(feat_low, xb_low, 256, 3136, 3136, BB);
  gap_pool_k<<<g1(32 * 2048), 256, 0, stream>>>(feat_out, poolT);

  // ---- 2) WMMA GEMMs: low projection + ASPP branches ----
  gemm(wb_low,  xb_low, low_raw, 48, 3136, 256, 3136, BB);
  gemm(wb_a0,   xb_out, b0_raw, 256, 196, 2048, 224, BB);
  conv3(wb_a1,  xb_out, b1_raw, 256, 2048, 14, 12, (long long)224 * 2048); // dil=12
  gemm(wb_a2c,  xb_out, b2_raw, 256, 196, 2048, 224, BB);   // degenerate 1x1
  gemm(wb_a3c,  xb_out, b3_raw, 256, 196, 2048, 224, BB);   // degenerate 1x1
  gemm(wb_pool, poolT,  gp_raw, 256, BB, 2048, 32, 1);

  // ---- 3) BatchNorm+ReLU into concat / ffea layouts ----
  bn(low_raw, g_low, b_low, 48, 3136, ffea_f, 304, 0, ffea_b, 320, 3136, 0);
  bn(b0_raw, g_a0, b_a0, 256, 196, nullptr, 0, 0, concat_b, 1280, 224, 0);
  bn(b1_raw, g_a1, b_a1, 256, 196, nullptr, 0, 0, concat_b, 1280, 224, 256);
  bn(b2_raw, g_a2, b_a2, 256, 196, nullptr, 0, 0, concat_b, 1280, 224, 512);
  bn(b3_raw, g_a3, b_a3, 256, 196, nullptr, 0, 0, concat_b, 1280, 224, 768);
  gp_bn_broadcast_k<<<1, 256, 0, stream>>>(gp_raw, g_pool, b_pool, concat_b);

  // ---- 4) ASPP projection + upsample into ffea ----
  gemm(wb_proj, concat_b, proj_raw, 256, 196, 1280, 224, BB);
  bn(proj_raw, g_proj, b_proj, 256, 196, aspp_act, 256, 0, nullptr, 1, 1, 0);
  resize14to56_k<<<g1(BB * 256 * 3136), 256, 0, stream>>>(aspp_act, ffea_f, ffea_b);

  // ---- 5) seg head ----
  conv3(wb_cls1, ffea_b, seg_raw, 256, 320, 56, 1, (long long)3136 * 320);
  bn(seg_raw, g_cls1, b_cls1, 256, 3136, nullptr, 0, 0, segact_b, 256, 3136, 0);
  gemm(wb_cls2, segact_b, cls2_raw, 1, 3136, 256, 3136, BB);
  bias_add_k<<<g1(BB * 3136), 256, 0, stream>>>(cls2_raw, cls2_bias, 1, 3136,
                                                BB * 3136, (float*)d_out);

  // ---- 6) recon head ----
  conv3(wb_rec1, ffea_b, rec_raw, 256, 320, 56, 1, (long long)3136 * 320);
  bn(rec_raw, g_rec1, b_rec1, 256, 3136, nullptr, 0, 0, recact_b, 256, 3136, 0);
  gemm(wb_rec2, recact_b, rec2_raw, 3, 3136, 256, 3136, BB);
  bias_add_k<<<g1(BB * 3 * 3136), 256, 0, stream>>>(rec2_raw, rec2_bias, 3, 3136,
                                                    BB * 3 * 3136, (float*)d_out + 12544);

  // ---- 7) fused resize(56->224) + segment mean + linear ----
  sp_accum_k<<<g1(BB * SPHW * SPHW), 256, 0, stream>>>(ffea_f, spmap, sums, counts);
  local_cls_k<<<dim3(BB * SPK), 64, 0, stream>>>(sums, counts, lin_w, lin_b,
                                                 (float*)d_out + 50176);
}
